// KmeansVectorQuantizer_35218731828027
// MI455X (gfx1250) — matrix operand order repaired and checked by reading, not compiled, not run on hardware
//
#include <hip/hip_runtime.h>

// ---------------------------------------------------------------------------
// Types for CDNA5 WMMA / TDM (wave32)
// ---------------------------------------------------------------------------
typedef __attribute__((ext_vector_type(16))) __bf16 v16bf;
typedef __attribute__((ext_vector_type(8)))  __bf16 v8bf;
typedef __attribute__((ext_vector_type(8)))  float  v8f;
typedef __attribute__((ext_vector_type(4)))  unsigned int v4u;
typedef __attribute__((ext_vector_type(8)))  int v8i;
typedef __attribute__((ext_vector_type(4)))  int v4i;

#define B_DIM 8
#define T_DIM 2048
#define C_DIM 1024
#define V_DIM 1024
#define NROWS (B_DIM * T_DIM)          // 16384
#define GN_EPS 1e-5f
#define GAMMA 0.25f

#define KC  128                        // k-chunk staged per TDM transfer
#define NKC (C_DIM / KC)               // 8 chunks

// ---------------------------------------------------------------------------
// Fragment loaders (contiguous k = hb*16 + s order, same for A and B, wave32, 16-bit 16x32 A / 32x16 B)
//   A (16x32, MxK): lanes 0-15 -> M=lane, K base 0; lanes 16-31 -> K base 16.
//     lane m holds 16 contiguous K values (K base = hb*16), matching B
//   B (32x16, KxN): lane n holds 16 contiguous K values (K base = hb*16)
// Works for both global (VMEM) and LDS (DS) source pointers.
// ---------------------------------------------------------------------------
__device__ __forceinline__ v16bf load_a_frag(const __bf16* base, int ld,
                                             int row0, int k0, int lane) {
  const int hb = lane >> 4;
  const int m  = lane & 15;
  const __bf16* p = base + (size_t)(row0 + m) * ld + k0 + hb * 16;
  v8bf lo = *(const v8bf*)p;
  v8bf hi = *(const v8bf*)(p + 8);
  v16bf r;
#pragma unroll
  for (int i = 0; i < 8; ++i) { r[i] = lo[i]; r[i + 8] = hi[i]; }
  return r;
}

__device__ __forceinline__ v16bf load_b_frag(const __bf16* base, int ld,
                                             int col0, int k0, int lane) {
  const int hb = lane >> 4;
  const int n  = lane & 15;
  return *(const v16bf*)(base + (size_t)(col0 + n) * ld + k0 + hb * 16);
}

// ---------------------------------------------------------------------------
// Tensor Data Mover: DMA a 2-D bf16 tile (tile1 rows x tile0 elems) from a
// row-major global matrix into LDS at byte offset lds_off.
// D# built per CDNA5 ISA ch.8: group0 = {count, lds_addr, global_addr, type=2},
// group1 = {data_size=2B, tensor dims, tile dims, dim0 stride}. Groups 2/3
// and the trailing group are zero (2-D tensor, no iterate/gather).
// Issued once per wave (EXEC ignored by TDM); tracked by TENSORcnt.
// NOTE: this toolchain exposes the 6-arg builtin
//   (uint32x4, int32x8, int32x4, int32x4, int32x8, i32 cpol).
// ---------------------------------------------------------------------------
__device__ __forceinline__ void tdm_load_tile_bf16(
    unsigned lds_off, const __bf16* gsrc,
    unsigned tdim0, unsigned tdim1, unsigned stride0,
    unsigned tile0, unsigned tile1) {
  unsigned long long ga = (unsigned long long)(size_t)gsrc;
  v4u g0;
  g0[0] = 1u;                                       // count=1, user mode
  g0[1] = lds_off;                                  // lds_addr (bytes)
  g0[2] = (unsigned)(ga & 0xFFFFFFFFull);           // global_addr[31:0]
  g0[3] = (unsigned)((ga >> 32) & 0x01FFFFFFull)    // global_addr[56:32]
          | 0x80000000u;                            // type=2 -> bits[127:126]=10b
  v8i g1;
  g1[0] = (int)(1u << 16);                          // data_size=1 (2 bytes)
  g1[1] = (int)(tdim0 << 16);                       // tensor_dim0[15:0]
  g1[2] = (int)((tdim0 >> 16) | (tdim1 << 16));     // tensor_dim0 hi | dim1 lo
  g1[3] = (int)((tdim1 >> 16) | (tile0 << 16));     // tensor_dim1 hi | tile_dim0
  g1[4] = (int)tile1;                               // tile_dim1 (tile_dim2=0)
  g1[5] = (int)stride0;                             // tensor_dim0_stride[31:0]
  g1[6] = 0;
  g1[7] = 0;
  v4i gz4 = {0, 0, 0, 0};
  v8i gz8 = {0, 0, 0, 0, 0, 0, 0, 0};
  __builtin_amdgcn_tensor_load_to_lds(g0, g1, gz4, gz4, gz8, 0);
}

// ---------------------------------------------------------------------------
// Kernel: zero the atomic accumulators
// ---------------------------------------------------------------------------
__global__ void zero_kernel(float* p, int n) {
  int i = blockIdx.x * blockDim.x + threadIdx.x;
  if (i < n) p[i] = 0.0f;
}

// ---------------------------------------------------------------------------
// Kernel: fp32 -> bf16 stream convert
// ---------------------------------------------------------------------------
__global__ void cvt_bf16_kernel(const float* __restrict__ src,
                                __bf16* __restrict__ dst, size_t n) {
  size_t i = (size_t)blockIdx.x * blockDim.x + threadIdx.x;
  size_t stride = (size_t)gridDim.x * blockDim.x;
  for (; i < n; i += stride) dst[i] = (__bf16)src[i];
}

// ---------------------------------------------------------------------------
// Kernel: embedding prep -> bf16 copy + per-codeword ||e||^2
// ---------------------------------------------------------------------------
__global__ __launch_bounds__(256) void emb_prep_kernel(
    const float* __restrict__ emb, __bf16* __restrict__ embb,
    float* __restrict__ embnorm) {
  const int v = blockIdx.x;
  __shared__ float red[256];
  float s = 0.f;
  for (int c = threadIdx.x; c < C_DIM; c += 256) {
    float e = emb[(size_t)v * C_DIM + c];
    embb[(size_t)v * C_DIM + c] = (__bf16)e;
    s += e * e;
  }
  red[threadIdx.x] = s;
  __syncthreads();
  for (int o = 128; o > 0; o >>= 1) {
    if (threadIdx.x < o) red[threadIdx.x] += red[threadIdx.x + o];
    __syncthreads();
  }
  if (threadIdx.x == 0) embnorm[v] = red[0];
}

// ---------------------------------------------------------------------------
// Kernel: GEMM1  ze[n,o] = sum_c xb[n,c] * wb[o,c]
// TDM double-buffered pipeline: wave 0 DMAs 64xKC bf16 A and B tiles into LDS
// (tensor_load_to_lds, TENSORcnt), all 8 waves compute WMMA from ds_loads
// while the next chunk streams in. Only dynamic LDS is used so the TDM
// descriptor lds_addr equals the byte offset into the dynamic segment.
// LDS: A[2][64*KC] + B[2][64*KC] bf16 (64KB) + 2KB reductions.
// ---------------------------------------------------------------------------
__global__ __launch_bounds__(256) void gemm1_kernel(
    const __bf16* __restrict__ xb, const __bf16* __restrict__ wb,
    float* __restrict__ ze, float* __restrict__ bsum) {
  extern __shared__ char smem[];
  const unsigned TILE_B = 64u * KC * 2u;            // 16 KB per tile
  const unsigned aOff[2] = {0u, TILE_B};
  const unsigned bOff[2] = {2u * TILE_B, 3u * TILE_B};
  float* rs  = (float*)(smem + 4u * TILE_B);
  float* rs2 = rs + 256;

  const int tid  = threadIdx.x;
  const int wave = tid >> 5;
  const int lane = tid & 31;
  const int row0g = blockIdx.x * 64;                // 256 m-blocks
  const int col0g = blockIdx.y * 64;                // 16  n-blocks
  const int msub  = (wave & 3) * 16;
  const int nsub  = (wave >> 2) * 16;

  // prologue: stage chunk 0
  if (wave == 0) {
    tdm_load_tile_bf16(aOff[0], xb + (size_t)row0g * C_DIM,
                       C_DIM, NROWS, C_DIM, KC, 64);
    tdm_load_tile_bf16(bOff[0], wb + (size_t)col0g * C_DIM,
                       C_DIM, C_DIM, C_DIM, KC, 64);
  }

  v8f acc0 = {}; v8f acc1 = {};
  for (int kc = 0; kc < NKC; ++kc) {
    if (wave == 0) __builtin_amdgcn_s_wait_tensorcnt(0);
    __syncthreads();  // chunk kc resident; everyone done with buffer kc^1
    const int cur = kc & 1;
    if (wave == 0 && kc + 1 < NKC) {
      const int k0n = (kc + 1) * KC;
      tdm_load_tile_bf16(aOff[cur ^ 1], xb + (size_t)row0g * C_DIM + k0n,
                         C_DIM, NROWS, C_DIM, KC, 64);
      tdm_load_tile_bf16(bOff[cur ^ 1], wb + (size_t)col0g * C_DIM + k0n,
                         C_DIM, C_DIM, C_DIM, KC, 64);
    }
    const __bf16* aB = (const __bf16*)(smem + aOff[cur]);
    const __bf16* bB = (const __bf16*)(smem + bOff[cur]);
#pragma unroll
    for (int kk = 0; kk < KC; kk += 32) {
      v16bf a  = load_a_frag(aB, KC, msub, kk, lane);
      v16bf b0 = load_b_frag(bB, KC, nsub, kk, lane);
      v16bf b1 = load_b_frag(bB, KC, nsub + 32, kk, lane);
      acc0 = __builtin_amdgcn_wmma_f32_16x16x32_bf16(false, a, false, b0,
                                                     (short)0, acc0, false, false);
      acc1 = __builtin_amdgcn_wmma_f32_16x16x32_bf16(false, a, false, b1,
                                                     (short)0, acc1, false, false);
    }
    __syncthreads();  // all reads of buffer `cur` done before it is refilled
  }

  // D layout: VGPR r, lanes 0-15 -> M=r ; lanes 16-31 -> M=8+r ; N=lane&15
  const int mofs = (lane >> 4) * 8;
  const int ncol = lane & 15;
  const int n0 = col0g + nsub;
  const int n1 = n0 + 32;
  float s = 0.f, s2 = 0.f;
#pragma unroll
  for (int r = 0; r < 8; ++r) {
    size_t row = (size_t)row0g + msub + mofs + r;
    float v0 = acc0[r], v1 = acc1[r];
    ze[row * C_DIM + n0 + ncol] = v0;
    ze[row * C_DIM + n1 + ncol] = v1;
    s += v0 + v1;
    s2 += v0 * v0 + v1 * v1;
  }

  rs[tid] = s; rs2[tid] = s2;
  __syncthreads();
  for (int o = 128; o > 0; o >>= 1) {
    if (tid < o) { rs[tid] += rs[tid + o]; rs2[tid] += rs2[tid + o]; }
    __syncthreads();
  }
  if (tid == 0) {
    int b = (blockIdx.x * 64) >> 11;  // 64-row tile never crosses a batch
    atomicAdd(&bsum[2 * b], rs[0]);
    atomicAdd(&bsum[2 * b + 1], rs2[0]);
  }
}

// ---------------------------------------------------------------------------
// Kernel: per-batch mean / rstd
// ---------------------------------------------------------------------------
__global__ void gn_stats_kernel(const float* __restrict__ bsum,
                                float* __restrict__ mstat) {
  int b = threadIdx.x;
  if (b < B_DIM) {
    const float n = (float)C_DIM * (float)T_DIM;
    float mean = bsum[2 * b] / n;
    float var  = bsum[2 * b + 1] / n - mean * mean;
    mstat[2 * b]     = mean;
    mstat[2 * b + 1] = rsqrtf(var + GN_EPS);
  }
}

// ---------------------------------------------------------------------------
// Kernel: normalize -> zn (bf16 for GEMM2) + per-row ||zn||^2
// ---------------------------------------------------------------------------
__global__ __launch_bounds__(256) void normalize_kernel(
    const float* __restrict__ ze, const float* __restrict__ mstat,
    const float* __restrict__ gnw, const float* __restrict__ gnb,
    __bf16* __restrict__ znb, float* __restrict__ rownorm) {
  const int row = blockIdx.x;
  const int b = row >> 11;  // row / T_DIM
  const float mean = mstat[2 * b], rstd = mstat[2 * b + 1];
  __shared__ float red[256];
  float s = 0.f;
  for (int c = threadIdx.x; c < C_DIM; c += 256) {
    float z = (ze[(size_t)row * C_DIM + c] - mean) * rstd * gnw[c] + gnb[c];
    znb[(size_t)row * C_DIM + c] = (__bf16)z;
    s += z * z;
  }
  red[threadIdx.x] = s;
  __syncthreads();
  for (int o = 128; o > 0; o >>= 1) {
    if (threadIdx.x < o) red[threadIdx.x] += red[threadIdx.x + o];
    __syncthreads();
  }
  if (threadIdx.x == 0) rownorm[row] = red[0];
}

// ---------------------------------------------------------------------------
// Kernel: fused GEMM2 + distance + softmax + argmin + gather + loss/counts.
// Block = 256 thr (8 waves) owns 64 rows; the full 64x1024 fp32 distance
// tile (256 KB) lives in dynamic LDS -- only possible with CDNA5's 320KB WGP LDS.
// emb (2 MB bf16) is L2-resident across all 256 blocks -> direct VMEM loads.
// ---------------------------------------------------------------------------
__global__ __launch_bounds__(256) void vq_kernel(
    const __bf16* __restrict__ znb, const __bf16* __restrict__ embb,
    const float* __restrict__ rownorm, const float* __restrict__ embnorm,
    const float* __restrict__ ze, const float* __restrict__ mstat,
    const float* __restrict__ gnw, const float* __restrict__ gnb,
    const float* __restrict__ embf,
    float* __restrict__ out_x, float* __restrict__ out_prob,
    float* __restrict__ out_targets,
    float* __restrict__ counts, float* __restrict__ loss_acc) {
  extern __shared__ float dmat[];  // [64][V_DIM] distances
  __shared__ float r_min[64][4];
  __shared__ int   r_idx[64][4];
  __shared__ float r_sum[64][4];
  __shared__ float rowmin[64];
  __shared__ int   rowarg[64];
  __shared__ float rowz[64];
  __shared__ float lossred[256];

  const int tid  = threadIdx.x;
  const int wave = tid >> 5;
  const int lane = tid & 31;
  const int row0 = blockIdx.x * 64;
  const int msub = (wave & 3) * 16;

  // ---- phase 1: dots via WMMA, convert to Euclidean distance, park in LDS
  for (int nt = (wave >> 2); nt < V_DIM / 16; nt += 2) {
    const int n0 = nt * 16;
    v8f acc = {};
#pragma unroll 4
    for (int k0 = 0; k0 < C_DIM; k0 += 32) {
      if (k0 + 32 < C_DIM) {
        __builtin_prefetch(znb + (size_t)(row0 + msub + (lane & 15)) * C_DIM + k0 + 32, 0, 3);
        __builtin_prefetch(embb + (size_t)(n0 + (lane & 15)) * C_DIM + k0 + 32, 0, 3);
      }
      v16bf a = load_a_frag(znb, C_DIM, row0 + msub, k0, lane);
      v16bf b = load_b_frag(embb, C_DIM, n0, k0, lane);
      acc = __builtin_amdgcn_wmma_f32_16x16x32_bf16(false, a, false, b,
                                                    (short)0, acc, false, false);
    }
    const int mofs = (lane >> 4) * 8;
    const int ncol = lane & 15;
    const float en = embnorm[n0 + ncol];
#pragma unroll
    for (int r = 0; r < 8; ++r) {
      int ml = msub + mofs + r;
      float sq = rownorm[row0 + ml] + en - 2.f * acc[r];
      dmat[ml * V_DIM + n0 + ncol] = sqrtf(fmaxf(sq, 0.f));
    }
  }
  __syncthreads();

  // ---- phase 2: per-row argmin + softmax(-d).  4 threads per row.
  const int row = tid >> 2;
  const int sub = tid & 3;
  float mn = 3.4e38f; int am = 0;
  for (int v = sub; v < V_DIM; v += 4) {
    float d = dmat[row * V_DIM + v];
    if (d < mn) { mn = d; am = v; }
  }
  r_min[row][sub] = mn; r_idx[row][sub] = am;
  __syncthreads();
  if (sub == 0) {
    float bmv = r_min[row][0]; int bi = r_idx[row][0];
#pragma unroll
    for (int j = 1; j < 4; ++j) {
      float m2 = r_min[row][j]; int i2 = r_idx[row][j];
      if (m2 < bmv || (m2 == bmv && i2 < bi)) { bmv = m2; bi = i2; }
    }
    rowmin[row] = bmv; rowarg[row] = bi;
  }
  __syncthreads();
  const float bm = rowmin[row];
  float se = 0.f;
  for (int v = sub; v < V_DIM; v += 4) se += __expf(bm - dmat[row * V_DIM + v]);
  r_sum[row][sub] = se;
  __syncthreads();
  if (sub == 0)
    rowz[row] = r_sum[row][0] + r_sum[row][1] + r_sum[row][2] + r_sum[row][3];
  __syncthreads();

  const float inv = 1.f / rowz[row];
  const size_t grow = (size_t)row0 + row;
  for (int v = sub; v < V_DIM; v += 4)
    out_prob[grow * V_DIM + v] = __expf(bm - dmat[row * V_DIM + v]) * inv;

  // ---- phase 3: gather codeword, commitment loss, counts, targets
  const int idx = rowarg[row];
  const int b = (int)(grow >> 11);
  const float mean = mstat[2 * b], rstd = mstat[2 * b + 1];
  float lp = 0.f;
  for (int c = sub; c < C_DIM; c += 4) {
    float e = embf[(size_t)idx * C_DIM + c];
    out_x[grow * C_DIM + c] = e;  // straight-through: out == emb[idx]
    float z = (ze[grow * C_DIM + c] - mean) * rstd * gnw[c] + gnb[c];
    float df = z - e;
    lp += df * df;
  }
  lossred[tid] = lp;
  __syncthreads();
  for (int o = 128; o > 0; o >>= 1) {
    if (tid < o) lossred[tid] += lossred[tid + o];
    __syncthreads();
  }
  if (tid == 0) atomicAdd(loss_acc, lossred[0]);
  if (sub == 0) {
    atomicAdd(&counts[idx], 1.0f);
    out_targets[grow] = (float)idx;
  }
}

// ---------------------------------------------------------------------------
// Kernel: perplexity + loss scalars
// ---------------------------------------------------------------------------
__global__ __launch_bounds__(256) void finalize_kernel(
    const float* __restrict__ counts, const float* __restrict__ loss_acc,
    float* __restrict__ out_cpx, float* __restrict__ out_loss) {
  __shared__ float red[256];
  float s = 0.f;
  const float invN = 1.0f / (float)NROWS;
  for (int v = threadIdx.x; v < V_DIM; v += 256) {
    float p = counts[v] * invN;
    s += p * __logf(p + 1e-7f);
  }
  red[threadIdx.x] = s;
  __syncthreads();
  for (int o = 128; o > 0; o >>= 1) {
    if (threadIdx.x < o) red[threadIdx.x] += red[threadIdx.x + o];
    __syncthreads();
  }
  if (threadIdx.x == 0) {
    *out_cpx  = __expf(-red[0]);
    *out_loss = GAMMA * loss_acc[0] / ((float)NROWS * (float)C_DIM);
  }
}

// ---------------------------------------------------------------------------
// Host-side orchestration (graph-capture safe: kernel launches only)
// ---------------------------------------------------------------------------
extern "C" void kernel_launch(void* const* d_in, const int* in_sizes, int n_in,
                              void* d_out, int out_size, void* d_ws, size_t ws_size,
                              hipStream_t stream) {
  const float* x     = (const float*)d_in[0];  // (B,T,C) -> rows n=b*T+t, contiguous c
  const float* convw = (const float*)d_in[1];  // (C,C)   -> rows o, contiguous c
  const float* gnw   = (const float*)d_in[2];
  const float* gnb   = (const float*)d_in[3];
  const float* embf  = (const float*)d_in[4];  // (V,1,C) -> rows v, contiguous c

  float* outf = (float*)d_out;
  float* out_x       = outf;                               // 16777216
  float* out_prob    = outf + (size_t)NROWS * C_DIM;       // 16777216
  float* out_cpx     = out_prob + (size_t)NROWS * V_DIM;   // 1
  float* out_loss    = out_cpx + 1;                        // 1
  float* out_targets = out_loss + 1;                       // 16384

  // ---- workspace layout (bytes) ----
  char* ws = (char*)d_ws;
  const size_t ZE_OFF   = 0;                                   // 64 MB f32
  const size_t ZNB_OFF  = ZE_OFF  + (size_t)NROWS * C_DIM * 4; // 32 MB bf16
  const size_t XB_OFF   = ZNB_OFF + (size_t)NROWS * C_DIM * 2; // 32 MB bf16
  const size_t WB_OFF   = XB_OFF  + (size_t)NROWS * C_DIM * 2; // 2 MB bf16
  const size_t EMBB_OFF = WB_OFF  + (size_t)C_DIM * C_DIM * 2; // 2 MB bf16
  const size_t EMBN_OFF = EMBB_OFF + (size_t)V_DIM * C_DIM * 2;
  const size_t ROWN_OFF = EMBN_OFF + (size_t)V_DIM * 4;
  const size_t STAT_OFF = ROWN_OFF + (size_t)NROWS * 4;
  float*  ze      = (float*)(ws + ZE_OFF);
  __bf16* znb     = (__bf16*)(ws + ZNB_OFF);
  __bf16* xb      = (__bf16*)(ws + XB_OFF);
  __bf16* wb      = (__bf16*)(ws + WB_OFF);
  __bf16* embb    = (__bf16*)(ws + EMBB_OFF);
  float*  embnorm = (float*)(ws + EMBN_OFF);
  float*  rownorm = (float*)(ws + ROWN_OFF);
  float*  bsum    = (float*)(ws + STAT_OFF);        // 16 f32
  float*  mstat   = bsum + 16;                      // 16 f32
  float*  counts  = mstat + 16;                     // 1024 f32
  float*  lossacc = counts + V_DIM;                 // 1 f32

  // 1) zero atomic accumulators (bsum+mstat+counts+loss = 1057 floats)
  zero_kernel<<<5, 256, 0, stream>>>(bsum, 16 + 16 + V_DIM + 1);

  // 2) fp32 -> bf16 conversions
  cvt_bf16_kernel<<<4096, 256, 0, stream>>>(x, xb, (size_t)NROWS * C_DIM);
  cvt_bf16_kernel<<<1024, 256, 0, stream>>>(convw, wb, (size_t)C_DIM * C_DIM);
  emb_prep_kernel<<<V_DIM, 256, 0, stream>>>(embf, embb, embnorm);

  // 3) GEMM1 (TDM-staged LDS + WMMA bf16 -> f32) + per-batch sum/sumsq
  dim3 g1(NROWS / 64, C_DIM / 64);
  const size_t g1_lds = 4u * (64u * KC * 2u) + 512u * sizeof(float); // 66 KB
  gemm1_kernel<<<g1, 256, g1_lds, stream>>>(xb, wb, ze, bsum);

  // 4) GroupNorm stats
  gn_stats_kernel<<<1, 32, 0, stream>>>(bsum, mstat);

  // 5) normalize -> zn bf16 + row norms
  normalize_kernel<<<NROWS, 256, 0, stream>>>(ze, mstat, gnw, gnb, znb, rownorm);

  // 6) fused GEMM2 + distance + softmax + argmin + gather + loss
  const size_t dmat_bytes = (size_t)64 * V_DIM * sizeof(float);  // 256 KB LDS
  vq_kernel<<<NROWS / 64, 256, dmat_bytes, stream>>>(
      znb, embb, rownorm, embnorm, ze, mstat, gnw, gnb, embf,
      out_x, out_prob, out_targets, counts, lossacc);

  // 7) perplexity + loss scalars
  finalize_kernel<<<1, 256, 0, stream>>>(counts, lossacc, out_cpx, out_loss);
}
